// decodeLayer_64716567216722
// MI455X (gfx1250) — matrix operand order repaired and checked by reading, not compiled, not run on hardware
//
#include <hip/hip_runtime.h>
#include <hip/hip_bf16.h>
#include <math.h>

typedef _Float16 f16;
typedef __attribute__((ext_vector_type(16))) _Float16 v16h;
typedef __attribute__((ext_vector_type(8)))  _Float16 v8h;
typedef __attribute__((ext_vector_type(8)))  float    v8f;
typedef __attribute__((ext_vector_type(4)))  unsigned int u32x4;
typedef __attribute__((ext_vector_type(8)))  int i32x8;
typedef __attribute__((ext_vector_type(4)))  int i32x4;

#define E_DIM 384
#define NH    6
#define HS    64
#define BB    64
#define TT    256
#define FF_DIM 1536
#define NTOK  (BB * TT)   // 16384

union Frag16 { v16h v; v8h h[2]; f16 e[16]; };

static __device__ __forceinline__ v8f wmma_f16(const Frag16& a, const Frag16& b, v8f c) {
  // D = A(16x32 f16) * B(32x16 f16) + C(16x16 f32)
  return __builtin_amdgcn_wmma_f32_16x16x32_f16(false, a.v, false, b.v, (short)0, c, false, false);
}

// ---------------------------------------------------------------------------
// f32 -> f16 conversions
// ---------------------------------------------------------------------------
__global__ void dec_cvt_f32_f16(const float* __restrict__ in, f16* __restrict__ out, int n) {
  int i = blockIdx.x * 256 + threadIdx.x;
  if (i < n) out[i] = (f16)in[i];
}

// [H, E, HS] f32 -> [E, H*HS] f16 (head-major columns, so QKV become plain GEMMs)
__global__ void dec_cvt_headw(const float* __restrict__ in, f16* __restrict__ out) {
  int i = blockIdx.x * 256 + threadIdx.x;
  if (i < NH * E_DIM * HS) {
    int h   = i / (E_DIM * HS);
    int rem = i % (E_DIM * HS);
    int e   = rem / HS;
    int s2  = rem % HS;
    out[e * (NH * HS) + h * HS + s2] = (f16)in[i];
  }
}

// ---------------------------------------------------------------------------
// Tiled WMMA GEMM: C[M,N] = A[M,K](f16) * B[K,N](f16) + bias, optional relu.
// Block tile 128x128, BK=32; 256 threads = 8 waves in a 2(m) x 4(n) grid;
// each wave owns a 64x32 sub-tile = 4x2 WMMA tiles (8 v_wmma per K-step).
// ---------------------------------------------------------------------------
#define BM 128
#define BN 128
#define BK 32
#define LDT (BK + 8)   // padded LDS row (80B, 16B aligned)

__global__ __launch_bounds__(256)
void dec_gemm_wmma(const f16* __restrict__ A, const f16* __restrict__ Bw,
                   const float* __restrict__ bias,
                   float* __restrict__ Cf, f16* __restrict__ Ch,
                   int M, int N, int K, int relu)
{
  __shared__ f16 As[BM][LDT];   // row-major M x K tile          (10 KB)
  __shared__ f16 Bs[BN][LDT];   // transposed [n][k] B tile      (10 KB)
  const int tid  = threadIdx.x;
  const int wave = tid >> 5, lane = tid & 31;
  const int hl   = lane >> 4, lm = lane & 15;
  const int wm   = wave >> 2;       // 0..1  -> 64 rows each
  const int wn   = wave & 3;        // 0..3  -> 32 cols each
  const int bM   = blockIdx.y * BM, bN = blockIdx.x * BN;

  v8f acc[4][2] = {};

  // per-thread source pointers for the cooperative tile loads
  const int am = tid >> 1, ac = (tid & 1) * 16;          // A: 16 halves each
  const int bk = tid >> 3, bn0 = (tid & 7) * 16;         // B: 16 halves each

  for (int kb = 0; kb < K; kb += BK) {
    __syncthreads();
    {   // A tile: 128x32 halves
      const f16* s = A + (size_t)(bM + am) * K + kb + ac;
      *(v8h*)&As[am][ac]     = *(const v8h*)s;
      *(v8h*)&As[am][ac + 8] = *(const v8h*)(s + 8);
    }
    {   // B tile: 32x128 halves, load contiguous, scatter transposed
      const f16* s = Bw + (size_t)(kb + bk) * N + bN + bn0;
      v8h t0 = *(const v8h*)s;
      v8h t1 = *(const v8h*)(s + 8);
      #pragma unroll
      for (int i = 0; i < 8; ++i) Bs[bn0 + i][bk]     = t0[i];
      #pragma unroll
      for (int i = 0; i < 8; ++i) Bs[bn0 + 8 + i][bk] = t1[i];
    }
    if (kb + BK < K) {   // prefetch next K-tile (global_prefetch_b8)
      __builtin_prefetch(A + (size_t)(bM + am) * K + kb + BK + ac, 0, 1);
      __builtin_prefetch(Bw + (size_t)(kb + BK + bk) * N + bN + bn0, 0, 1);
    }
    __syncthreads();

    Frag16 a[4], b[2];
    #pragma unroll
    for (int mt = 0; mt < 4; ++mt) {
      // A layout: lane%16=row, slots 0-7 -> K=hl*8+s, slots 8-15 -> K=16+hl*8+(s-8)
      const f16* r = &As[wm * 64 + mt * 16 + lm][0];
      a[mt].h[0] = *(const v8h*)(r + hl * 8);
      a[mt].h[1] = *(const v8h*)(r + 16 + hl * 8);
    }
    #pragma unroll
    for (int nt = 0; nt < 2; ++nt) {
      // B layout: lane%16=col, slots 0-7 -> K=hl*8+s, slots 8-15 -> K=16+hl*8+(s-8)
      const f16* r = &Bs[wn * 32 + nt * 16 + lm][0];
      b[nt].h[0] = *(const v8h*)(r + hl * 8);
      b[nt].h[1] = *(const v8h*)(r + 16 + hl * 8);
    }
    #pragma unroll
    for (int mt = 0; mt < 4; ++mt)
      #pragma unroll
      for (int nt = 0; nt < 2; ++nt)
        acc[mt][nt] = wmma_f16(a[mt], b[nt], acc[mt][nt]);
  }

  // Epilogue. C layout: reg r -> m = 8*hl + r, n = lane%16
  #pragma unroll
  for (int mt = 0; mt < 4; ++mt)
    #pragma unroll
    for (int nt = 0; nt < 2; ++nt) {
      int n = bN + wn * 32 + nt * 16 + lm;
      float bv = bias ? bias[n] : 0.f;
      #pragma unroll
      for (int r = 0; r < 8; ++r) {
        int m = bM + wm * 64 + mt * 16 + hl * 8 + r;
        float v = acc[mt][nt][r] + bv;
        if (relu) v = v > 0.f ? v : 0.f;
        if (Cf) Cf[(size_t)m * N + n] = v;
        if (Ch) Ch[(size_t)m * N + n] = (f16)v;
      }
    }
}

// ---------------------------------------------------------------------------
// Attention for one (b,h): S = Q K^T * scale, causal mask, softmax, O = P V.
// V tile staged into LDS by the Tensor Data Mover (TDM) when available.
// ---------------------------------------------------------------------------
__global__ __launch_bounds__(256)
void dec_attn_wmma(const f16* __restrict__ Q, const f16* __restrict__ K,
                   const f16* __restrict__ V, f16* __restrict__ O, float scale)
{
  __shared__ f16 Vs[TT][HS + 8];       // 36864 B (72-half padded rows)
  __shared__ f16 Ps[8][16][BK + 8];    // 10240 B  per-wave P restage buffer
  const int tid  = threadIdx.x;
  const int wave = tid >> 5, lane = tid & 31;
  const int hl   = lane >> 4, lm = lane & 15;
  const int b    = blockIdx.x / NH, h = blockIdx.x % NH;
  const size_t rowBase = (size_t)b * TT;

#if __has_builtin(__builtin_amdgcn_tensor_load_to_lds)
  // --- TDM path: DMA the 256x64 f16 V tile (row stride 384) into padded LDS ---
  if (wave == 0) {
    unsigned long long gaddr =
        (unsigned long long)(V + rowBase * E_DIM + (size_t)h * HS);
    unsigned int ldsoff = (unsigned int)(unsigned long long)(&Vs[0][0]);
    u32x4 g0;
    g0[0] = 1u;                                        // count=1, user mode
    g0[1] = ldsoff;                                    // lds_addr
    g0[2] = (unsigned int)(gaddr & 0xffffffffu);       // global_addr[31:0]
    g0[3] = (unsigned int)((gaddr >> 32) & 0x01ffffffu) | (2u << 30); // [56:32] | type=2
    i32x8 g1;
    // data_size=2B(code 1), pad_enable, pad_interval=32 DWORDs(code 4),
    // pad_amount=4 DWORDs(code 3)  -> 128B rows padded to 144B (72 halves)
    g1[0] = (int)((1u << 16) | (1u << 20) | (4u << 22) | (3u << 25));
    g1[1] = (int)((unsigned)E_DIM << 16);              // tensor_dim0 = 384 (low16 @ bits63:48)
    g1[2] = (int)(((unsigned)NTOK & 0xffffu) << 16);   // dim0 hi16=0 | tensor_dim1=16384 low16
    g1[3] = (int)((unsigned)HS << 16);                 // dim1 hi16=0 | tile_dim0 = 64
    g1[4] = (int)TT;                                   // tile_dim1 = 256, tile_dim2 = 0
    g1[5] = (int)E_DIM;                                // tensor_dim0_stride = 384 (low 32)
    g1[6] = 0;                                         // stride hi | tensor_dim1_stride lo
    g1[7] = 0;
    i32x4 gz4 = {0, 0, 0, 0};
    i32x8 gz8 = {0, 0, 0, 0, 0, 0, 0, 0};
    // 6-arg form on this toolchain: (g0, g1, g2, g3, g4, cpol)
    __builtin_amdgcn_tensor_load_to_lds(g0, g1, gz4, gz4, gz8, 0);
    __builtin_amdgcn_s_wait_tensorcnt(0);
  }
#else
  {   // fallback: vector-load staging, one V row per thread
    const f16* s = V + (rowBase + tid) * E_DIM + h * HS;
    #pragma unroll
    for (int i = 0; i < 8; ++i)
      *(v8h*)&Vs[tid][i * 8] = *(const v8h*)(s + i * 8);
  }
#endif
  __syncthreads();

  #pragma unroll 1
  for (int rb = 0; rb < 2; ++rb) {
    const int m0 = wave * 32 + rb * 16;

    Frag16 aQ[2];
    {
      const f16* qr = Q + (rowBase + m0 + lm) * E_DIM + h * HS;
      #pragma unroll
      for (int kc = 0; kc < 2; ++kc) {
        aQ[kc].h[0] = *(const v8h*)(qr + kc * 32 + hl * 8);
        aQ[kc].h[1] = *(const v8h*)(qr + kc * 32 + 16 + hl * 8);
      }
    }

    v8f S[16] = {};
    #pragma unroll 1
    for (int j = 0; j < 16; ++j) {
      #pragma unroll
      for (int kc = 0; kc < 2; ++kc) {
        const f16* kr = K + (rowBase + j * 16 + lm) * E_DIM + h * HS + kc * 32 + hl * 8;
        Frag16 bk;
        bk.h[0] = *(const v8h*)kr;
        bk.h[1] = *(const v8h*)(kr + 16);
        S[j] = wmma_f16(aQ[kc], bk, S[j]);
      }
    }

    // scale + causal mask + row softmax. Row m lives in 16 lanes of one half.
    #pragma unroll
    for (int r = 0; r < 8; ++r) {
      const int tq = m0 + hl * 8 + r;
      float mx = -3.0e38f;
      #pragma unroll
      for (int j = 0; j < 16; ++j) {
        int u = j * 16 + lm;
        float v = (u <= tq) ? S[j][r] * scale : -3.0e38f;
        S[j][r] = v;
        mx = fmaxf(mx, v);
      }
      #pragma unroll
      for (int d = 1; d < 16; d <<= 1) mx = fmaxf(mx, __shfl_xor(mx, d, 32));
      float sum = 0.f;
      #pragma unroll
      for (int j = 0; j < 16; ++j) {
        float e = __expf(S[j][r] - mx);
        S[j][r] = e;
        sum += e;
      }
      #pragma unroll
      for (int d = 1; d < 16; d <<= 1) sum += __shfl_xor(sum, d, 32);
      float inv = 1.f / sum;
      #pragma unroll
      for (int j = 0; j < 16; ++j) S[j][r] *= inv;
    }

    // O[16,64] = P[16,256] * V[256,64]
    v8f o[4] = {};
    #pragma unroll 1
    for (int t = 0; t < 8; ++t) {
      // restage P tile pair (C layout f32 -> A layout f16) via wave-private LDS
      #pragma unroll
      for (int r = 0; r < 8; ++r) {
        Ps[wave][hl * 8 + r][lm]      = (f16)S[2 * t][r];
        Ps[wave][hl * 8 + r][16 + lm] = (f16)S[2 * t + 1][r];
      }
      asm volatile("s_wait_dscnt 0x0" ::: "memory");  // same-wave LDS ordering
      Frag16 ap;
      ap.h[0] = *(const v8h*)&Ps[wave][lm][hl * 8];
      ap.h[1] = *(const v8h*)&Ps[wave][lm][16 + hl * 8];
      #pragma unroll
      for (int nt = 0; nt < 4; ++nt) {
        Frag16 bv;
        #pragma unroll
        for (int s2 = 0; s2 < 16; ++s2)
          bv.e[s2] = Vs[t * 32 + hl * 8 + (s2 & 7) + (s2 >> 3) * 16][nt * 16 + lm];
        o[nt] = wmma_f16(ap, bv, o[nt]);
      }
    }

    #pragma unroll
    for (int nt = 0; nt < 4; ++nt)
      #pragma unroll
      for (int r = 0; r < 8; ++r)
        O[(rowBase + m0 + hl * 8 + r) * E_DIM + h * HS + nt * 16 + lm] = (f16)o[nt][r];
  }
}

// ---------------------------------------------------------------------------
// out = LayerNorm(x + a) * g + beta ; writes f32 and (optionally) f16 copy.
// ---------------------------------------------------------------------------
__global__ __launch_bounds__(128)
void dec_add_ln(const float* __restrict__ X, const float* __restrict__ A,
                const float* __restrict__ g, const float* __restrict__ beta,
                float* __restrict__ Of, f16* __restrict__ Oh)
{
  __shared__ float red[4];
  const int row = blockIdx.x, tid = threadIdx.x;
  const int wave = tid >> 5, lane = tid & 31;
  const float* x = X + (size_t)row * E_DIM;
  const float* a = A + (size_t)row * E_DIM;

  float v[3];
  #pragma unroll
  for (int i = 0; i < 3; ++i) v[i] = x[tid + i * 128] + a[tid + i * 128];

  float s = v[0] + v[1] + v[2];
  #pragma unroll
  for (int dd = 1; dd < 32; dd <<= 1) s += __shfl_xor(s, dd, 32);
  if (lane == 0) red[wave] = s;
  __syncthreads();
  float mean = (red[0] + red[1] + red[2] + red[3]) * (1.0f / E_DIM);
  __syncthreads();

  float q = 0.f;
  #pragma unroll
  for (int i = 0; i < 3; ++i) { float d = v[i] - mean; q += d * d; }
  #pragma unroll
  for (int dd = 1; dd < 32; dd <<= 1) q += __shfl_xor(q, dd, 32);
  if (lane == 0) red[wave] = q;
  __syncthreads();
  float var = (red[0] + red[1] + red[2] + red[3]) * (1.0f / E_DIM);
  float rstd = rsqrtf(var + 1e-5f);

  #pragma unroll
  for (int i = 0; i < 3; ++i) {
    int c = tid + i * 128;
    float o = (v[i] - mean) * rstd * g[c] + beta[c];
    Of[(size_t)row * E_DIM + c] = o;
    if (Oh) Oh[(size_t)row * E_DIM + c] = (f16)o;
  }
}

// ---------------------------------------------------------------------------
// Host orchestration
// ---------------------------------------------------------------------------
extern "C" void kernel_launch(void* const* d_in, const int* in_sizes, int n_in,
                              void* d_out, int out_size, void* d_ws, size_t ws_size,
                              hipStream_t stream) {
  const float* enc  = (const float*)d_in[0];
  const float* x    = (const float*)d_in[1];
  const float* wq1  = (const float*)d_in[2];
  const float* bq1  = (const float*)d_in[3];
  const float* wk1  = (const float*)d_in[4];
  const float* bk1  = (const float*)d_in[5];
  const float* wv1  = (const float*)d_in[6];
  const float* bv1  = (const float*)d_in[7];
  const float* pw1  = (const float*)d_in[8];
  const float* pb1  = (const float*)d_in[9];
  const float* wq2  = (const float*)d_in[10];
  const float* bq2  = (const float*)d_in[11];
  const float* wk2  = (const float*)d_in[12];
  const float* bk2  = (const float*)d_in[13];
  const float* wv2  = (const float*)d_in[14];
  const float* bv2  = (const float*)d_in[15];
  const float* pw2  = (const float*)d_in[16];
  const float* pb2  = (const float*)d_in[17];
  const float* ffw1 = (const float*)d_in[18];
  const float* ffb1 = (const float*)d_in[19];
  const float* ffw2 = (const float*)d_in[20];
  const float* ffb2 = (const float*)d_in[21];
  const float* ln1g = (const float*)d_in[22];
  const float* ln1b = (const float*)d_in[23];
  const float* ln2g = (const float*)d_in[24];
  const float* ln2b = (const float*)d_in[25];
  const float* ln3g = (const float*)d_in[26];
  const float* ln3b = (const float*)d_in[27];

  char* ws = (char*)d_ws;
  size_t off = 0;
  auto carve = [&](size_t bytes) -> void* {
    void* p = ws + off;
    off += (bytes + 255) & ~(size_t)255;
    return p;
  };

  const size_t actH16 = (size_t)NTOK * E_DIM * sizeof(f16);   // 12.6 MB
  const size_t actF32 = (size_t)NTOK * E_DIM * sizeof(float); // 25.2 MB
  f16*   xh   = (f16*)carve(actH16);
  f16*   eh   = (f16*)carve(actH16);
  f16*   Qh   = (f16*)carve(actH16);
  f16*   Kh   = (f16*)carve(actH16);
  f16*   Vh   = (f16*)carve(actH16);
  f16*   Oh   = (f16*)carve(actH16);
  f16*   aH   = (f16*)carve(actH16);                           // o1h then o2h
  float* pre  = (float*)carve(actF32);                         // proj / ffn pre-LN
  float* res1 = (float*)carve(actF32);                         // o1 (f32)
  float* res2 = (float*)carve(actF32);                         // o2 (f32)
  f16*   f1h  = (f16*)carve((size_t)NTOK * FF_DIM * sizeof(f16));
  const size_t wEE = (size_t)E_DIM * E_DIM * sizeof(f16);
  f16* wq1h = (f16*)carve(wEE); f16* wk1h = (f16*)carve(wEE); f16* wv1h = (f16*)carve(wEE);
  f16* wp1h = (f16*)carve(wEE);
  f16* wq2h = (f16*)carve(wEE); f16* wk2h = (f16*)carve(wEE); f16* wv2h = (f16*)carve(wEE);
  f16* wp2h = (f16*)carve(wEE);
  f16* wf1h = (f16*)carve((size_t)E_DIM * FF_DIM * sizeof(f16));
  f16* wf2h = (f16*)carve((size_t)FF_DIM * E_DIM * sizeof(f16));
  (void)ws_size; (void)n_in; (void)in_sizes; (void)out_size;

  const int nAct = NTOK * E_DIM;
  const int nHW  = NH * E_DIM * HS;
  const float scale = 0.05103103630798288f;  // 384^-0.5 (reference uses q_in last dim!)

  dec_cvt_f32_f16<<<(nAct + 255) / 256, 256, 0, stream>>>(x, xh, nAct);
  dec_cvt_f32_f16<<<(nAct + 255) / 256, 256, 0, stream>>>(enc, eh, nAct);
  dec_cvt_headw<<<(nHW + 255) / 256, 256, 0, stream>>>(wq1, wq1h);
  dec_cvt_headw<<<(nHW + 255) / 256, 256, 0, stream>>>(wk1, wk1h);
  dec_cvt_headw<<<(nHW + 255) / 256, 256, 0, stream>>>(wv1, wv1h);
  dec_cvt_headw<<<(nHW + 255) / 256, 256, 0, stream>>>(wq2, wq2h);
  dec_cvt_headw<<<(nHW + 255) / 256, 256, 0, stream>>>(wk2, wk2h);
  dec_cvt_headw<<<(nHW + 255) / 256, 256, 0, stream>>>(wv2, wv2h);
  const int nEE = E_DIM * E_DIM, nEF = E_DIM * FF_DIM;
  dec_cvt_f32_f16<<<(nEE + 255) / 256, 256, 0, stream>>>(pw1, wp1h, nEE);
  dec_cvt_f32_f16<<<(nEE + 255) / 256, 256, 0, stream>>>(pw2, wp2h, nEE);
  dec_cvt_f32_f16<<<(nEF + 255) / 256, 256, 0, stream>>>(ffw1, wf1h, nEF);
  dec_cvt_f32_f16<<<(nEF + 255) / 256, 256, 0, stream>>>(ffw2, wf2h, nEF);

  dim3 gE(E_DIM / BN, NTOK / BM);   // (3,128)
  dim3 gF(FF_DIM / BN, NTOK / BM);  // (12,128)

  // ---- self-attention block ----
  dec_gemm_wmma<<<gE, 256, 0, stream>>>(xh, wq1h, bq1, nullptr, Qh, NTOK, E_DIM, E_DIM, 0);
  dec_gemm_wmma<<<gE, 256, 0, stream>>>(xh, wk1h, bk1, nullptr, Kh, NTOK, E_DIM, E_DIM, 0);
  dec_gemm_wmma<<<gE, 256, 0, stream>>>(xh, wv1h, bv1, nullptr, Vh, NTOK, E_DIM, E_DIM, 0);
  dec_attn_wmma<<<BB * NH, 256, 0, stream>>>(Qh, Kh, Vh, Oh, scale);
  dec_gemm_wmma<<<gE, 256, 0, stream>>>(Oh, wp1h, pb1, pre, nullptr, NTOK, E_DIM, E_DIM, 0);
  dec_add_ln<<<NTOK, 128, 0, stream>>>(x, pre, ln1g, ln1b, res1, aH);

  // ---- cross-attention block (q = o1, k/v = encode_output; mask still applied) ----
  dec_gemm_wmma<<<gE, 256, 0, stream>>>(aH, wq2h, bq2, nullptr, Qh, NTOK, E_DIM, E_DIM, 0);
  dec_gemm_wmma<<<gE, 256, 0, stream>>>(eh, wk2h, bk2, nullptr, Kh, NTOK, E_DIM, E_DIM, 0);
  dec_gemm_wmma<<<gE, 256, 0, stream>>>(eh, wv2h, bv2, nullptr, Vh, NTOK, E_DIM, E_DIM, 0);
  dec_attn_wmma<<<BB * NH, 256, 0, stream>>>(Qh, Kh, Vh, Oh, scale);
  dec_gemm_wmma<<<gE, 256, 0, stream>>>(Oh, wp2h, pb2, pre, nullptr, NTOK, E_DIM, E_DIM, 0);
  dec_add_ln<<<NTOK, 128, 0, stream>>>(res1, pre, ln2g, ln2b, res2, aH);

  // ---- FFN block ----
  dec_gemm_wmma<<<gF, 256, 0, stream>>>(aH, wf1h, ffb1, nullptr, f1h, NTOK, FF_DIM, E_DIM, 1);
  dec_gemm_wmma<<<gE, 256, 0, stream>>>(f1h, wf2h, ffb2, pre, nullptr, NTOK, E_DIM, FF_DIM, 0);
  dec_add_ln<<<NTOK, 128, 0, stream>>>(res2, pre, ln3g, ln3b, (float*)d_out, nullptr);
}
